// CalculateHLayer_58471684767747
// MI455X (gfx1250) — matrix operand order repaired and checked by reading, not compiled, not run on hardware
//
#include <hip/hip_runtime.h>

// ---------------------------------------------------------------------------
// CalculateHLayer for MI455X (gfx1250, wave32, WMMA)
//
// A = adj[:,:,0]+adj[:,:,1]  (f32 -> f16, one pass over adj = 134 MB)
// h_in  = A^T @ h   (f32 accumulate via v_wmma_f32_16x16x32_f16)
// h_out = A   @ h
//
// Roofline: 10.1 GFLOP / 134 MB ~ 75 FLOP/B; with f16 WMMA the kernel is
// HBM-streaming bound (~5.8 us at 23.3 TB/s), so we read adj exactly once
// and derive both matmuls from the same LDS-resident tile (+ its transpose).
// ---------------------------------------------------------------------------

static constexpr int Ln   = 4096;  // nodes
static constexpr int Dd   = 150;   // feature dim
static constexpr int BT   = 128;   // tile edge (i and j)
static constexpr int NP   = 160;   // Dd padded to multiple of 16
static constexpr int AST  = 136;   // A-tile stride (halves): 128+8 pad, 16B-aligned rows, conflict-free frags
static constexpr int HST  = 136;   // H-tile stride (halves): k-dim 128+8 pad

typedef _Float16 v8h  __attribute__((ext_vector_type(8)));
typedef _Float16 v16h __attribute__((ext_vector_type(16)));
typedef float    v8f  __attribute__((ext_vector_type(8)));

static __device__ __forceinline__ v16h cat16(v8h lo, v8h hi) {
  return __builtin_shufflevector(lo, hi, 0,1,2,3,4,5,6,7,8,9,10,11,12,13,14,15);
}

// A-operand fragment (16x32 f16) from row-major tile [m][k], stride AST halves.
// ISA layout: lane L holds M=L%16; halves hv: K = (hv/8)*16 + (L/16)*8 + hv%8
// -> two contiguous 16B chunks per lane (ds_load_b128 x2).
static __device__ __forceinline__ v16h ldsA(const _Float16* t, int m0, int kc, int lane) {
  const int off = (m0 + (lane & 15)) * AST + kc * 32 + ((lane >> 4) << 3);
  v8h lo = *(const v8h*)(t + off);
  v8h hi = *(const v8h*)(t + off + 16);
  return cat16(lo, hi);
}

// B-operand fragment (32x16 f16) from tile stored [n][k], stride HST halves.
// ISA layout: lane L holds N=L%16; half hv: K = (hv/8)*16 + (L/16)*8 + hv%8
// -> two 8-half runs per lane, 16 halves apart (two aligned 16B chunks).
static __device__ __forceinline__ v16h ldsB(const _Float16* t, int n0, int kc, int lane) {
  const int off = (n0 + (lane & 15)) * HST + kc * 32 + ((lane >> 4) << 3);
  v8h lo = *(const v8h*)(t + off);
  v8h hi = *(const v8h*)(t + off + 16);
  return cat16(lo, hi);
}

static __device__ __forceinline__ v8f wmma_f16(v16h a, v16h b, v8f c) {
  // (neg_a, A, neg_b, B, c_mod, C, reuse_a, reuse_b)
  return __builtin_amdgcn_wmma_f32_16x16x32_f16(false, a, false, b, (short)0, c, false, false);
}

__global__ void __launch_bounds__(256)
zero_f32_kernel(float* __restrict__ p, int n) {
  const int i = blockIdx.x * blockDim.x + threadIdx.x;
  if (i < n) p[i] = 0.0f;
}

__global__ void __launch_bounds__(256)
calc_h_layer_kernel(const float* __restrict__ adj,
                    const float* __restrict__ h,
                    float* __restrict__ out) {
  extern __shared__ char smem[];
  _Float16* Af  = (_Float16*)smem;          // [BT][AST]  A tile, row-major (M=i, K=j)
  _Float16* ATf = Af  + BT * AST;           // [BT][AST]  A^T tile        (M=j, K=i)
  _Float16* Hjf = ATf + BT * AST;           // [NP][HST]  h[j0..j0+127]^T ([n][k])
  _Float16* Hif = Hjf + NP * HST;           // [NP][HST]  h[i0..i0+127]^T ([n][k])

  const int tid = threadIdx.x;
  const int i0  = blockIdx.x * BT;
  const int j0  = blockIdx.y * BT;

  // ---- Stage adj tile: sum 2 edge channels in f32, convert once to f16,
  //      write both orientations. Each float4 = {adj[i][j][0..1], adj[i][j+1][0..1]}.
  for (int it = 0; it < 32; ++it) {
    const int q   = tid + 256 * it;      // 0..8191 = 128 rows x 64 float4
    const int row = q >> 6;
    const int jl  = (q & 63) << 1;
    const float4 v = *(const float4*)(adj + ((size_t)(i0 + row) * Ln + (j0 + jl)) * 2);
    const _Float16 a0 = (_Float16)(v.x + v.y);
    const _Float16 a1 = (_Float16)(v.z + v.w);
    Af[row * AST + jl]        = a0;
    Af[row * AST + jl + 1]    = a1;
    ATf[jl * AST + row]       = a0;
    ATf[(jl + 1) * AST + row] = a1;
  }

  // ---- Stage h tiles column-major [n][k] (B-operand friendly), zero-pad n in [150,160)
  for (int p = tid; p < NP * BT; p += 256) {
    const int n = p % NP;
    const int k = p / NP;
    const float vj = (n < Dd) ? h[(size_t)(j0 + k) * Dd + n] : 0.0f;
    const float vi = (n < Dd) ? h[(size_t)(i0 + k) * Dd + n] : 0.0f;
    Hjf[n * HST + k] = (_Float16)vj;
    Hif[n * HST + k] = (_Float16)vi;
  }
  __syncthreads();

  const int lane = tid & 31;
  const int wid  = tid >> 5;
  const int wm   = wid & 3;    // M group: rows 32*wm..32*wm+31 (2 WMMA M-tiles)
  const int wn   = wid >> 2;   // N group: n-tiles 5*wn..5*wn+4

  v8f co[2][5] = {};           // h_out accumulators (A   @ Hj)
  v8f ci[2][5] = {};           // h_in  accumulators (A^T @ Hi)

  for (int kc = 0; kc < 4; ++kc) {       // K in chunks of 32
    const v16h a0 = ldsA(Af,  32 * wm,      kc, lane);
    const v16h a1 = ldsA(Af,  32 * wm + 16, kc, lane);
    const v16h t0 = ldsA(ATf, 32 * wm,      kc, lane);
    const v16h t1 = ldsA(ATf, 32 * wm + 16, kc, lane);
#pragma unroll
    for (int nt = 0; nt < 5; ++nt) {
      const int n0 = (5 * wn + nt) << 4;
      const v16h bj = ldsB(Hjf, n0, kc, lane);
      co[0][nt] = wmma_f16(a0, bj, co[0][nt]);
      co[1][nt] = wmma_f16(a1, bj, co[1][nt]);
      const v16h bi = ldsB(Hif, n0, kc, lane);
      ci[0][nt] = wmma_f16(t0, bi, ci[0][nt]);
      ci[1][nt] = wmma_f16(t1, bi, ci[1][nt]);
    }
  }

  // ---- Accumulate partial tiles into L2-resident outputs.
  // C/D layout: VGPR r, lane L -> M = r + 8*(L/16), N = L%16.
  float* hin  = out;                       // h_in  first in tuple order
  float* hout = out + (size_t)Ln * Dd;     // then h_out
  const int rbase = (lane >> 4) << 3;
  const int ncol  = lane & 15;
#pragma unroll
  for (int mt = 0; mt < 2; ++mt) {
#pragma unroll
    for (int nt = 0; nt < 5; ++nt) {
      const int dcol = ((5 * wn + nt) << 4) + ncol;
      if (dcol < Dd) {
#pragma unroll
        for (int r = 0; r < 8; ++r) {
          const int gr = 32 * wm + 16 * mt + rbase + r;
          unsafeAtomicAdd(&hout[(size_t)(i0 + gr) * Dd + dcol], co[mt][nt][r]);
          unsafeAtomicAdd(&hin [(size_t)(j0 + gr) * Dd + dcol], ci[mt][nt][r]);
        }
      }
    }
  }
}

extern "C" void kernel_launch(void* const* d_in, const int* in_sizes, int n_in,
                              void* d_out, int out_size, void* d_ws, size_t ws_size,
                              hipStream_t stream) {
  (void)in_sizes; (void)n_in; (void)d_ws; (void)ws_size;
  const float* adj = (const float*)d_in[0];  // [4096,4096,2] f32
  const float* h   = (const float*)d_in[1];  // [4096,150]    f32
  float* out = (float*)d_out;                // h_in ++ h_out, f32

  // d_out is poisoned by the harness; zero it before atomic accumulation.
  zero_f32_kernel<<<(out_size + 255) / 256, 256, 0, stream>>>(out, out_size);

  constexpr size_t smem_bytes =
      (size_t)(2 * BT * AST + 2 * NP * HST) * sizeof(_Float16);  // 156,672 B < 320 KB/WGP
  calc_h_layer_kernel<<<dim3(Ln / BT, Ln / BT), 256, smem_bytes, stream>>>(adj, h, out);
}
